// SAKELayer_13108240187517
// MI455X (gfx1250) — matrix operand order repaired and checked by reading, not compiled, not run on hardware
//
#include <hip/hip_runtime.h>
#include <hip/hip_bf16.h>

typedef __attribute__((ext_vector_type(16))) _Float16 v16h;
typedef __attribute__((ext_vector_type(8)))  _Float16 v8h;
typedef __attribute__((ext_vector_type(8)))  float    v8f;

#define N_NODES 50000
#define DEG 10
#define FDIM 64
#define HDIM 64
#define SDIM 8

// ---- workspace layout (bytes); weight buffers are WMMA-B-fragment swizzled:
// [frag = kt*4+nt][lane 0..31][16 halves], 1 frag = 1024 B ----
#define O_DSUM 0
#define O_EM1  32                                   // 20 frags (K=160 padded from 137)
#define O_EM2  (O_EM1 + 20*1024)                    // 8 frags  (K=64)
#define O_CM1  (O_EM2 + 8*1024)                     // 8 frags
#define O_NM1  (O_CM1 + 8*1024)                     // 20 frags (K=160 padded from 136)
#define O_NM2  (O_NM1 + 20*1024)                    // 8 frags
#define O_HEDX (O_NM2 + 8*1024)                     // [E][8] f32
#define O_HVDX (O_HEDX + N_NODES*DEG*SDIM*4)        // [N][8] f32

__device__ __forceinline__ float siluf(float x) { return x / (1.0f + __expf(-x)); }

__device__ __forceinline__ v8h cvt8(const float4 a, const float4 b) {
  v8h h;
  h[0] = (_Float16)a.x; h[1] = (_Float16)a.y; h[2] = (_Float16)a.z; h[3] = (_Float16)a.w;
  h[4] = (_Float16)b.x; h[5] = (_Float16)b.y; h[6] = (_Float16)b.z; h[7] = (_Float16)b.w;
  return h;
}

// A-fragment (16x32 f16) from LDS, row-major [rows][ldk] halves.
// layout: m=lane&15; K=v+16*half (matches swizzled B fragments)
__device__ __forceinline__ v16h afrag_lds(const _Float16* S, int ldk, int row0, int k0, int lane) {
  const int m  = row0 + (lane & 15);
  const int h16 = (lane >> 4) << 4;
  v16h a;
#pragma unroll
  for (int v = 0; v < 16; ++v) {
    const int kk = v + h16;
    a[v] = S[m * ldk + k0 + kk];
  }
  return a;
}

// B-fragment: one contiguous 32-byte load from pre-swizzled weights
__device__ __forceinline__ v16h bfrag(const _Float16* Wf, int frag, int lane) {
  return *(const v16h*)(Wf + (frag * 32 + lane) * 16);
}

// ================= K0: zero dsum + f32 -> f16 B-fragment swizzle =================
__global__ __launch_bounds__(256)
void sake_prep_kernel(const float* __restrict__ em_w1, const float* __restrict__ em_w2,
                      const float* __restrict__ cm_w1, const float* __restrict__ nm_w1,
                      const float* __restrict__ nm_w2,
                      double* __restrict__ dsum,
                      _Float16* __restrict__ em1f, _Float16* __restrict__ em2f,
                      _Float16* __restrict__ cm1f, _Float16* __restrict__ nm1f,
                      _Float16* __restrict__ nm2f) {
  const int t0 = blockIdx.x * blockDim.x + threadIdx.x;
  const int stride = gridDim.x * blockDim.x;
  if (t0 == 0) *dsum = 0.0;
  // fragment element i -> (frag, lane, v) -> source (k, n)
  for (int i = t0; i < 20 * 512; i += stride) {
    const int v = i & 15, lane = (i >> 4) & 31, frag = i >> 9;
    const int kt = frag >> 2, nt = frag & 3;
    const int k = kt * 32 + ((lane >> 4) << 4) + v;
    const int n = nt * 16 + (lane & 15);
    em1f[i] = (_Float16)((k < 137) ? em_w1[k * 64 + n] : 0.0f);
    nm1f[i] = (_Float16)((k < 136) ? nm_w1[k * 64 + n] : 0.0f);
  }
  for (int i = t0; i < 8 * 512; i += stride) {
    const int v = i & 15, lane = (i >> 4) & 31, frag = i >> 9;
    const int kt = frag >> 2, nt = frag & 3;
    const int k = kt * 32 + ((lane >> 4) << 4) + v;
    const int n = nt * 16 + (lane & 15);
    em2f[i] = (_Float16)em_w2[k * 64 + n];
    cm1f[i] = (_Float16)cm_w1[k * 64 + n];
    nm2f[i] = (_Float16)nm_w2[k * 64 + n];
  }
}

// ================= K1: global sum of pairwise squared distances =================
__global__ __launch_bounds__(256)
void sake_deltasum_kernel(const float* __restrict__ coord, const int* __restrict__ src,
                          double* __restrict__ dsum) {
  __shared__ double red[256];
  const int node = blockIdx.x * 256 + threadIdx.x;
  float part = 0.0f;
  if (node < N_NODES) {
    float cx[DEG], cy[DEG], cz[DEG];
#pragma unroll
    for (int k = 0; k < DEG; ++k) {
      const int s = src[node * DEG + k];
      cx[k] = coord[s * 3 + 0];
      cy[k] = coord[s * 3 + 1];
      cz[k] = coord[s * 3 + 2];
    }
    float acc = 0.0f;
#pragma unroll
    for (int i = 0; i < DEG; ++i)
#pragma unroll
      for (int j = i + 1; j < DEG; ++j) {
        const float a = cx[i] - cx[j], b = cy[i] - cy[j], c = cz[i] - cz[j];
        acc += a * a + b * b + c * c;
      }
    part = 2.0f * acc;  // symmetric, zero diagonal
  }
  red[threadIdx.x] = (double)part;
  __syncthreads();
  for (int off = 128; off > 0; off >>= 1) {
    if (threadIdx.x < off) red[threadIdx.x] += red[threadIdx.x + off];
    __syncthreads();
  }
  if (threadIdx.x == 0) atomicAdd(dsum, red[0]);
}

// ================= K2: mailbox coordinate attention (1 wave = 1 node) =================
__global__ __launch_bounds__(256)
void sake_mailbox_kernel(const float* __restrict__ coord, const int* __restrict__ src,
                         const float* __restrict__ dcm_w1, const float* __restrict__ dcm_b1,
                         const float* __restrict__ dcm_w2, const float* __restrict__ dcm_b2,
                         const float* __restrict__ esm_w, const float* __restrict__ esm_b,
                         const float* __restrict__ nsm_w, const float* __restrict__ nsm_b,
                         const double* __restrict__ dsum,
                         float* __restrict__ hedx, float* __restrict__ hvdx) {
  __shared__ float sC[8 * DEG * 4];        // neighbor coords
  __shared__ float sH[8 * 100 * SDIM];     // h [D*D][S]
  __shared__ float sE[8 * DEG * 40];       // PNA over i -> [D][5S]
  __shared__ float sEd[8 * DEG * SDIM];    // h_e_dx [D][S]
  __shared__ float sV[8 * 40];             // PNA over j -> [5S]
  __shared__ float sWe[320], sWn[320];     // esm_w, nsm_w
  __shared__ float sDcm[88];               // dcm: w1[8] b1[8] w2[64] b2[8]

  const int wave = threadIdx.x >> 5;
  const int lane = threadIdx.x & 31;
  const int node = blockIdx.x * 8 + wave;
  const float inv = (float)(1.0 / (*dsum + 1.0));

  for (int i = threadIdx.x; i < 320; i += 256) { sWe[i] = esm_w[i]; sWn[i] = nsm_w[i]; }
  if (threadIdx.x < 88) {
    const int i = threadIdx.x;
    float v;
    if (i < 8) v = dcm_w1[i];
    else if (i < 16) v = dcm_b1[i - 8];
    else if (i < 80) v = dcm_w2[i - 16];
    else v = dcm_b2[i - 80];
    sDcm[i] = v;
  }
  if (lane < DEG) {
    const int s = src[node * DEG + lane];
    sC[(wave * DEG + lane) * 4 + 0] = coord[s * 3 + 0];
    sC[(wave * DEG + lane) * 4 + 1] = coord[s * 3 + 1];
    sC[(wave * DEG + lane) * 4 + 2] = coord[s * 3 + 2];
  }
  __syncthreads();

  // per-pair MLP: Linear(1->8) SiLU, Linear(8->8) SiLU
  for (int p = lane; p < 100; p += 32) {
    const int i = p / 10, j = p % 10;
    float dd = 0.0f;
#pragma unroll
    for (int ax = 0; ax < 3; ++ax) {
      const float t = sC[(wave * DEG + i) * 4 + ax] - sC[(wave * DEG + j) * 4 + ax];
      dd += t * t;
    }
    const float nd = dd * inv;
    float t1[SDIM];
#pragma unroll
    for (int s = 0; s < SDIM; ++s) t1[s] = siluf(nd * sDcm[s] + sDcm[8 + s]);
#pragma unroll
    for (int s = 0; s < SDIM; ++s) {
      float a = sDcm[80 + s];
#pragma unroll
      for (int u = 0; u < SDIM; ++u) a += t1[u] * sDcm[16 + u * SDIM + s];
      sH[(wave * 100 + p) * SDIM + s] = siluf(a);
    }
  }
  __syncthreads();

  // PNA over i for each (j,s): mean,sum,max,min,std(ddof=1)
  for (int idx = lane; idx < 80; idx += 32) {
    const int j = idx >> 3, s = idx & 7;
    float sum = 0.0f, sq = 0.0f, mx = -1e30f, mn = 1e30f;
#pragma unroll
    for (int i = 0; i < DEG; ++i) {
      const float v = sH[(wave * 100 + i * 10 + j) * SDIM + s];
      sum += v; sq += v * v;
      mx = fmaxf(mx, v); mn = fminf(mn, v);
    }
    const float mean = sum * 0.1f;
    float var = (sq - sum * mean) * (1.0f / 9.0f);
    var = fmaxf(var, 0.0f);
    float* e = &sE[(wave * DEG + j) * 40];
    e[0 + s] = mean; e[8 + s] = sum; e[16 + s] = mx; e[24 + s] = mn; e[32 + s] = sqrtf(var);
  }
  __syncthreads();

  // edge_summary: Linear(40->8) SiLU
  for (int idx = lane; idx < 80; idx += 32) {
    const int j = idx >> 3, s = idx & 7;
    float a = esm_b[s];
#pragma unroll
    for (int u = 0; u < 40; ++u) a += sE[(wave * DEG + j) * 40 + u] * sWe[u * SDIM + s];
    const float v = siluf(a);
    sEd[(wave * DEG + j) * SDIM + s] = v;
    hedx[(node * DEG + j) * SDIM + s] = v;
  }
  __syncthreads();

  // PNA over j, then node_summary: Linear(40->8) SiLU
  if (lane < SDIM) {
    const int s = lane;
    float sum = 0.0f, sq = 0.0f, mx = -1e30f, mn = 1e30f;
#pragma unroll
    for (int j = 0; j < DEG; ++j) {
      const float v = sEd[(wave * DEG + j) * SDIM + s];
      sum += v; sq += v * v;
      mx = fmaxf(mx, v); mn = fminf(mn, v);
    }
    const float mean = sum * 0.1f;
    float var = (sq - sum * mean) * (1.0f / 9.0f);
    var = fmaxf(var, 0.0f);
    float* vv = &sV[wave * 40];
    vv[0 + s] = mean; vv[8 + s] = sum; vv[16 + s] = mx; vv[24 + s] = mn; vv[32 + s] = sqrtf(var);
  }
  __syncthreads();
  if (lane < SDIM) {
    float a = nsm_b[lane];
#pragma unroll
    for (int u = 0; u < 40; ++u) a += sV[wave * 40 + u] * sWn[u * SDIM + lane];
    hvdx[node * SDIM + lane] = siluf(a);
  }
}

// ================= K3: fused edge MLP + coord MLP + aggregation + node MLP =================
// block = 16 nodes = 160 edges = 10 waves; each wave owns one 16-edge WMMA tile.
__global__ __launch_bounds__(320, 1)
void sake_edge_node_kernel(const float* __restrict__ feat, const float* __restrict__ coord,
                           const int* __restrict__ src,
                           const float* __restrict__ em_b1, const float* __restrict__ em_b2,
                           const float* __restrict__ cm_b1, const float* __restrict__ cm_w2,
                           const float* __restrict__ cm_b2,
                           const float* __restrict__ nm_b1, const float* __restrict__ nm_b2,
                           const _Float16* __restrict__ em1f, const _Float16* __restrict__ em2f,
                           const _Float16* __restrict__ cm1f, const _Float16* __restrict__ nm1f,
                           const _Float16* __restrict__ nm2f,
                           const float* __restrict__ hedx, const float* __restrict__ hvdx,
                           float* __restrict__ hout, float* __restrict__ xout) {
  // phase-aliased LDS: region A reused under barrier separation
  __shared__ __align__(16) unsigned char smem[51200 + 2560 + 640 + 640];
  _Float16* sIn  = (_Float16*)smem;                 // [160][160] e_in f16        (ph1-2)
  _Float16* sH1  = (_Float16*)smem;                 // [160][72]  h1 / h_e f16    (ph3-6)
  _Float16* sNI  = (_Float16*)(smem + 23040);       // [16][160]  node input f16  (ph6-7)
  _Float16* sN1  = (_Float16*)(smem + 28160);       // [16][72]   node hidden f16 (ph7)
  float*    sDx  = (float*)(smem + 51200);          // [160][4]   dx0 dx1 dx2 d2
  float*    sSc  = (float*)(smem + 53760);          // [160]      coord-MLP scalar
  int*      sSrc = (int*)(smem + 54400);            // [160]      src per edge

  const int tid  = threadIdx.x;
  const int wave = tid >> 5;
  const int lane = tid & 31;
  const int nn   = lane & 15;
  const int hx   = lane >> 4;
  const int n0   = blockIdx.x * 16;
  const int e0   = n0 * DEG;

  // ---- Phase 1a: per-edge src / dx / d2 into LDS ----
  if (hx == 0) {
    const int el = wave * 16 + nn;
    const int e  = e0 + el;
    const int s  = src[e];
    const int dn = n0 + el / DEG;
    const float dx0 = coord[s * 3 + 0] - coord[dn * 3 + 0];
    const float dx1 = coord[s * 3 + 1] - coord[dn * 3 + 1];
    const float dx2 = coord[s * 3 + 2] - coord[dn * 3 + 2];
    sDx[el * 4 + 0] = dx0; sDx[el * 4 + 1] = dx1; sDx[el * 4 + 2] = dx2;
    sDx[el * 4 + 3] = dx0 * dx0 + dx1 * dx1 + dx2 * dx2;
    sSrc[el] = s;
  }
  __syncthreads();

  // ---- Phase 1b: stage e_in = [h_e_dx(8) | feat[src](64) | feat[dst](64) | d2 | pad] ----
  // 16 edges x 20 octets (8 f16 each); all segment boundaries are octet-aligned.
#pragma unroll 1
  for (int it = 0; it < 10; ++it) {
    const int idx = it * 32 + lane;          // 0..319 within this wave's tile
    const int el  = wave * 16 + idx / 20;
    const int oc  = idx % 20;
    v8h h;
    if (oc < 17) {
      const float* p;
      if (oc == 0)      p = hedx + (e0 + el) * 8;
      else if (oc < 9)  p = feat + sSrc[el] * 64 + (oc - 1) * 8;
      else              p = feat + (n0 + el / DEG) * 64 + (oc - 9) * 8;
      if (oc == 16) {   // feat[dst][56..63] then d2 goes in octet 17
        h = cvt8(*(const float4*)p, *(const float4*)(p + 4));
      } else {
        h = cvt8(*(const float4*)p, *(const float4*)(p + 4));
      }
    } else {
#pragma unroll
      for (int q = 0; q < 8; ++q) h[q] = (_Float16)0.0f;
      if (oc == 17) h[0] = (_Float16)sDx[el * 4 + 3];   // d2 at feature 136
    }
    *(v8h*)(sIn + el * 160 + oc * 8) = h;
  }
  __syncthreads();

  v8f acc[4];
#pragma unroll
  for (int t = 0; t < 4; ++t)
#pragma unroll
    for (int i = 0; i < 8; ++i) acc[t][i] = 0.0f;

  // ---- Phase 2: edge layer 1  [16,160] x [160,64] ----
#pragma unroll
  for (int kt = 0; kt < 5; ++kt) {
    const v16h a = afrag_lds(sIn, 160, wave * 16, kt * 32, lane);
#pragma unroll
    for (int t = 0; t < 4; ++t) {
      const v16h b = bfrag(em1f, kt * 4 + t, lane);
      acc[t] = __builtin_amdgcn_wmma_f32_16x16x32_f16(false, a, false, b, (short)0, acc[t], false, false);
    }
  }
  __syncthreads();  // all waves done reading sIn before region A is overwritten

  // ---- Phase 3: bias + SiLU -> sH1 (f16) ----
#pragma unroll
  for (int t = 0; t < 4; ++t) {
    const float bb = em_b1[t * 16 + nn];
#pragma unroll
    for (int r = 0; r < 8; ++r) {
      const int m = wave * 16 + r + hx * 8;           // D layout: M = r + 8*half
      sH1[m * 72 + t * 16 + nn] = (_Float16)siluf(acc[t][r] + bb);
    }
  }
  __syncthreads();

  // ---- Phase 4: edge layer 2  [16,64] x [64,64]; h_e = SiLU(.) back into sH1 ----
#pragma unroll
  for (int t = 0; t < 4; ++t)
#pragma unroll
    for (int i = 0; i < 8; ++i) acc[t][i] = 0.0f;
#pragma unroll
  for (int kt = 0; kt < 2; ++kt) {
    const v16h a = afrag_lds(sH1, 72, wave * 16, kt * 32, lane);
#pragma unroll
    for (int t = 0; t < 4; ++t) {
      const v16h b = bfrag(em2f, kt * 4 + t, lane);
      acc[t] = __builtin_amdgcn_wmma_f32_16x16x32_f16(false, a, false, b, (short)0, acc[t], false, false);
    }
  }
  __syncthreads();
#pragma unroll
  for (int t = 0; t < 4; ++t) {
    const float bb = em_b2[t * 16 + nn];
#pragma unroll
    for (int r = 0; r < 8; ++r) {
      const int m = wave * 16 + r + hx * 8;
      sH1[m * 72 + t * 16 + nn] = (_Float16)siluf(acc[t][r] + bb);   // h_e (f16)
    }
  }
  __syncthreads();

  // ---- Phase 5: coord MLP: t = SiLU(h_e @ cm_w1 + b); scalar = t . cm_w2 + b2 ----
#pragma unroll
  for (int t = 0; t < 4; ++t)
#pragma unroll
    for (int i = 0; i < 8; ++i) acc[t][i] = 0.0f;
#pragma unroll
  for (int kt = 0; kt < 2; ++kt) {
    const v16h a = afrag_lds(sH1, 72, wave * 16, kt * 32, lane);
#pragma unroll
    for (int t = 0; t < 4; ++t) {
      const v16h b = bfrag(cm1f, kt * 4 + t, lane);
      acc[t] = __builtin_amdgcn_wmma_f32_16x16x32_f16(false, a, false, b, (short)0, acc[t], false, false);
    }
  }
  {
    float p[8];
#pragma unroll
    for (int r = 0; r < 8; ++r) p[r] = 0.0f;
#pragma unroll
    for (int t = 0; t < 4; ++t) {
      const float bb = cm_b1[t * 16 + nn];
      const float w2 = cm_w2[t * 16 + nn];
#pragma unroll
      for (int r = 0; r < 8; ++r) p[r] += siluf(acc[t][r] + bb) * w2;
    }
    // reduce over the 16 lanes (N index) within each half: lane-bits 0..3 butterfly
#pragma unroll
    for (int off = 1; off < 16; off <<= 1)
#pragma unroll
      for (int r = 0; r < 8; ++r) p[r] += __shfl_xor(p[r], off, 32);
    if (nn == 0) {
      const float b2 = cm_b2[0];
#pragma unroll
      for (int r = 0; r < 8; ++r) sSc[wave * 16 + r + hx * 8] = p[r] + b2;
    }
  }
  __syncthreads();

  // ---- Phase 6a: x_out = coord + sum_k dx*scal (10 contiguous edges / node) ----
  if (tid < 48) {
    const int ln = tid / 3, ax = tid % 3;
    float s = 0.0f;
#pragma unroll
    for (int k = 0; k < DEG; ++k) {
      const int el = ln * DEG + k;
      s += sDx[el * 4 + ax] * sSc[el];
    }
    const int node = n0 + ln;
    xout[node * 3 + ax] = coord[node * 3 + ax] + s;
  }
  // ---- Phase 6b: stage node input [feat | h_agg | h_v_dx | pad] as 16 x 20 octets ----
  {
    const int ln = tid / 20, oc = tid % 20;
    v8h h;
    if (oc < 8) {
      const float* p = feat + (n0 + ln) * 64 + oc * 8;
      h = cvt8(*(const float4*)p, *(const float4*)(p + 4));
    } else if (oc < 16) {
      const int ff = (oc - 8) * 8;
      float a8[8];
#pragma unroll
      for (int c = 0; c < 8; ++c) a8[c] = 0.0f;
#pragma unroll
      for (int k = 0; k < DEG; ++k) {
        const v8h t = *(const v8h*)(sH1 + (ln * DEG + k) * 72 + ff);
#pragma unroll
        for (int c = 0; c < 8; ++c) a8[c] += (float)t[c];
      }
#pragma unroll
      for (int c = 0; c < 8; ++c) h[c] = (_Float16)a8[c];   // h_agg
    } else if (oc == 16) {
      const float* p = hvdx + (n0 + ln) * 8;
      h = cvt8(*(const float4*)p, *(const float4*)(p + 4));
    } else {
#pragma unroll
      for (int c = 0; c < 8; ++c) h[c] = (_Float16)0.0f;
    }
    *(v8h*)(sNI + ln * 160 + oc * 8) = h;
  }
  __syncthreads();

  // ---- Phase 7: node MLP, waves 0..3 each own one 16-wide N tile ----
  if (wave < 4) {
    v8f an;
#pragma unroll
    for (int i = 0; i < 8; ++i) an[i] = 0.0f;
#pragma unroll
    for (int kt = 0; kt < 5; ++kt) {
      const v16h a = afrag_lds(sNI, 160, 0, kt * 32, lane);
      const v16h b = bfrag(nm1f, kt * 4 + wave, lane);
      an = __builtin_amdgcn_wmma_f32_16x16x32_f16(false, a, false, b, (short)0, an, false, false);
    }
    const float bb = nm_b1[wave * 16 + nn];
#pragma unroll
    for (int r = 0; r < 8; ++r)
      sN1[(r + hx * 8) * 72 + wave * 16 + nn] = (_Float16)siluf(an[r] + bb);
  }
  __syncthreads();
  if (wave < 4) {
    v8f an;
#pragma unroll
    for (int i = 0; i < 8; ++i) an[i] = 0.0f;
#pragma unroll
    for (int kt = 0; kt < 2; ++kt) {
      const v16h a = afrag_lds(sN1, 72, 0, kt * 32, lane);
      const v16h b = bfrag(nm2f, kt * 4 + wave, lane);
      an = __builtin_amdgcn_wmma_f32_16x16x32_f16(false, a, false, b, (short)0, an, false, false);
    }
    const float bb = nm_b2[wave * 16 + nn];
#pragma unroll
    for (int r = 0; r < 8; ++r) {
      const int m = r + hx * 8;
      hout[(n0 + m) * 64 + wave * 16 + nn] = an[r] + bb;   // no trailing SiLU
    }
  }
}

extern "C" void kernel_launch(void* const* d_in, const int* in_sizes, int n_in,
                              void* d_out, int out_size, void* d_ws, size_t ws_size,
                              hipStream_t stream) {
  const float* feat   = (const float*)d_in[0];
  const float* coord  = (const float*)d_in[1];
  const int*   src    = (const int*)d_in[2];
  const float* dcm_w1 = (const float*)d_in[3];
  const float* dcm_b1 = (const float*)d_in[4];
  const float* dcm_w2 = (const float*)d_in[5];
  const float* dcm_b2 = (const float*)d_in[6];
  const float* esm_w  = (const float*)d_in[7];
  const float* esm_b  = (const float*)d_in[8];
  const float* nsm_w  = (const float*)d_in[9];
  const float* nsm_b  = (const float*)d_in[10];
  const float* em_w1  = (const float*)d_in[11];
  const float* em_b1  = (const float*)d_in[12];
  const float* em_w2  = (const float*)d_in[13];
  const float* em_b2  = (const float*)d_in[14];
  const float* cm_w1  = (const float*)d_in[15];
  const float* cm_b1  = (const float*)d_in[16];
  const float* cm_w2  = (const float*)d_in[17];
  const float* cm_b2  = (const float*)d_in[18];
  const float* nm_w1  = (const float*)d_in[19];
  const float* nm_b1  = (const float*)d_in[20];
  const float* nm_w2  = (const float*)d_in[21];
  const float* nm_b2  = (const float*)d_in[22];

  unsigned char* ws = (unsigned char*)d_ws;
  double*    dsum = (double*)(ws + O_DSUM);
  _Float16*  em1f = (_Float16*)(ws + O_EM1);
  _Float16*  em2f = (_Float16*)(ws + O_EM2);
  _Float16*  cm1f = (_Float16*)(ws + O_CM1);
  _Float16*  nm1f = (_Float16*)(ws + O_NM1);
  _Float16*  nm2f = (_Float16*)(ws + O_NM2);
  float*     hedx = (float*)(ws + O_HEDX);
  float*     hvdx = (float*)(ws + O_HVDX);

  float* hout = (float*)d_out;                   // [N,64]
  float* xout = (float*)d_out + N_NODES * 64;    // [N,3]

  sake_prep_kernel<<<64, 256, 0, stream>>>(em_w1, em_w2, cm_w1, nm_w1, nm_w2,
                                           dsum, em1f, em2f, cm1f, nm1f, nm2f);
  sake_deltasum_kernel<<<(N_NODES + 255) / 256, 256, 0, stream>>>(coord, src, dsum);
  sake_mailbox_kernel<<<N_NODES / 8, 256, 0, stream>>>(coord, src,
                                                       dcm_w1, dcm_b1, dcm_w2, dcm_b2,
                                                       esm_w, esm_b, nsm_w, nsm_b,
                                                       dsum, hedx, hvdx);
  sake_edge_node_kernel<<<N_NODES / 16, 320, 0, stream>>>(feat, coord, src,
                                                          em_b1, em_b2, cm_b1, cm_w2, cm_b2,
                                                          nm_b1, nm_b2,
                                                          em1f, em2f, cm1f, nm1f, nm2f,
                                                          hedx, hvdx, hout, xout);
}
